// SparseConvNeXtV2Block3D_14310831031000
// MI455X (gfx1250) — matrix operand order repaired and checked by reading, not compiled, not run on hardware
//
#include <hip/hip_runtime.h>

// ---------------- problem constants ----------------
#define NPTS   100000
#define CDIM   64
#define KKOFF  343
#define TILES  (NPTS / 16)        // 6250
#define CONV_WAVES 16             // 512-thread blocks
#define CONV_UNITS (NPTS / 32)    // 3125 units of 32 rows
#define CONV_BLOCKS ((CONV_UNITS + CONV_WAVES - 1) / CONV_WAVES)   // 196
#define PW_WAVES 2                // 6250 = 3125 * 2
#define PW_BLOCKS (TILES / PW_WAVES)

// ---------------- workspace layout (bytes, 256-aligned) ----------------
#define OFF_FEATSBF 0UL                         // (N+1)*64 bf16 = 12,800,128 -> pad
#define OFF_WBDW    12800256UL                  // 343 * 8KB packed B frags
#define OFF_WB1     (OFF_WBDW + 2809856UL)      // 32 KB
#define OFF_WB2     (OFF_WB1 + 32768UL)         // 32 KB
#define OFF_X       (OFF_WB2 + 32768UL)         // N*64 f32 conv output
#define OFF_H       (OFF_X + 25600000UL)        // N*256 bf16 hidden
#define OFF_STATS   (OFF_H + 51200000UL)        // B*3 f32 (sum, sumsq, cnt)
#define OFF_GX2     (OFF_STATS + 256UL)         // B*256 f32
#define OFF_SCALE   (OFF_GX2 + 2048UL)          // B*256 f32

typedef __attribute__((ext_vector_type(16))) __bf16 v16bf;
typedef __attribute__((ext_vector_type(8)))  float  v8f;

union Frag {
  unsigned int u[8];
  uint4 q[2];
  v16bf v;
};

__device__ __forceinline__ unsigned short f2bf(float f) {
  unsigned int u = __float_as_uint(f);
  u += 0x7fffu + ((u >> 16) & 1u);           // round-to-nearest-even
  return (unsigned short)(u >> 16);
}
__device__ __forceinline__ unsigned int pack2bf(float a, float b) {
  return (unsigned int)f2bf(a) | ((unsigned int)f2bf(b) << 16);
}
__device__ __forceinline__ float bflo(unsigned int d) { return __uint_as_float(d << 16); }
__device__ __forceinline__ float bfhi(unsigned int d) { return __uint_as_float(d & 0xffff0000u); }

#define WMMA_BF16(A, B, C) \
  C = __builtin_amdgcn_wmma_f32_16x16x32_bf16(false, (A).v, false, (B).v, (short)0, C, false, false)

// ---------------- small init: zero stats, gx2, zero-row of featsbf ----------------
__global__ void init_kernel(float* stats, float* gx2, unsigned int* rowN) {
  int t = threadIdx.x;
  if (t < 6)   stats[t] = 0.f;
  if (t < 512) gx2[t]   = 0.f;
  if (t < 32)  rowN[t]  = 0u;   // 64 bf16 channels = 32 dwords
}

// ---------------- feats f32 -> bf16 (packed dwords) ----------------
__global__ void cvt_feats(const float* __restrict__ src, unsigned int* __restrict__ dst, int ndw) {
  int i = blockIdx.x * blockDim.x + threadIdx.x;
  if (i < ndw) dst[i] = pack2bf(src[2 * i], src[2 * i + 1]);
}

// ---------------- pack [K, Ncols] f32 weight into WMMA B fragments ----------------
// Fragment layout: frag = ntile*(Ktot/32) + kchunk ; inside: dword = lane*8 + v.
// Lane n holds column ntile*16 + (n&15); K = kchunk*32 + (n>=16)*8 + (v>=4)*16 + 2(v&3) (+1 in hi half).
__global__ void pack_w(const float* __restrict__ src, unsigned int* __restrict__ dst,
                       int Ktot, int Ncols, int total) {
  int id = blockIdx.x * blockDim.x + threadIdx.x;
  if (id >= total) return;
  int kch  = Ktot >> 5;
  int mdw  = kch * (Ncols >> 4) * 256;
  int mat  = id / mdw;
  int rem  = id % mdw;
  int frag = rem >> 8;
  int lr   = rem & 255;
  int lane = lr >> 3;
  int v    = lr & 7;
  int nt   = frag / kch;
  int kc   = frag % kch;
  int col  = nt * 16 + (lane & 15);
  int Klo  = kc * 32 + ((lane >> 4) << 3) + ((v >> 2) << 4) + 2 * (v & 3);
  const float* sm = src + (size_t)mat * Ktot * Ncols;
  dst[id] = pack2bf(sm[Klo * Ncols + col], sm[(Klo + 1) * Ncols + col]);
}

// ---------------- conv epilogue helper ----------------
__device__ __forceinline__ void store_conv(float* __restrict__ xout, const float* __restrict__ b_dw,
                                           v8f a, int rb, int nt, int colb) {
  float bs = b_dw[nt * 16 + colb];
#pragma unroll
  for (int j = 0; j < 8; ++j)
    xout[(size_t)(rb + j) * 64 + nt * 16 + colb] = a[j] + bs;
}

// ---------------- sparse 7x7x7 conv: 343 gathered 32x64x64 GEMMs per wave ----------------
__global__ __launch_bounds__(512) void conv_dw_wmma(
    const unsigned short* __restrict__ featsbf,   // [N+1, 64] bf16
    const uint4* __restrict__ wB,                 // packed: k * 512 uint4 (8KB each)
    const int* __restrict__ nbr,                  // [N, 343]
    const float* __restrict__ b_dw,
    float* __restrict__ xout)                     // [N, 64] f32
{
  __shared__ uint4 sB[2][512];                    // double-buffered 64x64 bf16 B
  const int tid  = threadIdx.x;
  const int lane = tid & 31;
  int unit = blockIdx.x * CONV_WAVES + (tid >> 5);
  if (unit >= CONV_UNITS) unit = CONV_UNITS - 1;  // padded waves duplicate last unit
  const int half = lane >> 4;
  const int r0 = unit * 32 + (lane & 15);         // tile X rows
  const int r1 = r0 + 16;                         // tile Y rows
  const int nb0 = r0 * KKOFF;
  const int nb1 = r1 * KKOFF;

  v8f ax0 = {}, ax1 = {}, ax2 = {}, ax3 = {};     // tile X accumulators (4 n-tiles)
  v8f ay0 = {}, ay1 = {}, ay2 = {}, ay3 = {};     // tile Y accumulators

  int idx0 = nbr[nb0];                            // software-pipelined neighbor indices
  int idx1 = nbr[nb1];
  for (int k = 0; k < KKOFF; ++k) {
    // --- A gathers for both tiles (issue early, overlap staging + barrier) ---
    const unsigned short* ar0 = featsbf + (size_t)idx0 * 64 + half * 8;
    const unsigned short* ar1 = featsbf + (size_t)idx1 * 64 + half * 8;
    Frag x0, x1, y0, y1;
    x0.q[0] = *(const uint4*)(ar0);
    x0.q[1] = *(const uint4*)(ar0 + 16);
    x1.q[0] = *(const uint4*)(ar0 + 32);
    x1.q[1] = *(const uint4*)(ar0 + 48);
    y0.q[0] = *(const uint4*)(ar1);
    y0.q[1] = *(const uint4*)(ar1 + 16);
    y1.q[0] = *(const uint4*)(ar1 + 32);
    y1.q[1] = *(const uint4*)(ar1 + 48);
    if (k + 1 < KKOFF) {
      idx0 = nbr[nb0 + k + 1];
      idx1 = nbr[nb1 + k + 1];
    }

    // --- stage B matrix for this k: exactly one uint4 per thread ---
    uint4* sb = sB[k & 1];
    sb[tid] = wB[(size_t)k * 512 + tid];
    if (k + 1 < KKOFF)
      __builtin_prefetch((const void*)(wB + (size_t)(k + 1) * 512 + tid), 0, 3);
    __syncthreads();

    // --- 8 B fragments from LDS, 16 WMMAs (each B frag feeds 2 tiles) ---
    const uint4* p = sb + lane * 2;
    Frag b0, b1, b2, b3, b4, b5, b6, b7;
    b0.q[0] = p[0 * 64]; b0.q[1] = p[0 * 64 + 1];
    b1.q[0] = p[1 * 64]; b1.q[1] = p[1 * 64 + 1];
    b2.q[0] = p[2 * 64]; b2.q[1] = p[2 * 64 + 1];
    b3.q[0] = p[3 * 64]; b3.q[1] = p[3 * 64 + 1];
    b4.q[0] = p[4 * 64]; b4.q[1] = p[4 * 64 + 1];
    b5.q[0] = p[5 * 64]; b5.q[1] = p[5 * 64 + 1];
    b6.q[0] = p[6 * 64]; b6.q[1] = p[6 * 64 + 1];
    b7.q[0] = p[7 * 64]; b7.q[1] = p[7 * 64 + 1];
    WMMA_BF16(x0, b0, ax0); WMMA_BF16(y0, b0, ay0);
    WMMA_BF16(x1, b1, ax0); WMMA_BF16(y1, b1, ay0);
    WMMA_BF16(x0, b2, ax1); WMMA_BF16(y0, b2, ay1);
    WMMA_BF16(x1, b3, ax1); WMMA_BF16(y1, b3, ay1);
    WMMA_BF16(x0, b4, ax2); WMMA_BF16(y0, b4, ay2);
    WMMA_BF16(x1, b5, ax2); WMMA_BF16(y1, b5, ay2);
    WMMA_BF16(x0, b6, ax3); WMMA_BF16(y0, b6, ay3);
    WMMA_BF16(x1, b7, ax3); WMMA_BF16(y1, b7, ay3);
    // no tail barrier: double buffering + the k-1 sync guarantees buffer safety
  }

  const int colb = lane & 15;
  const int rbx = unit * 32 + half * 8;
  const int rby = rbx + 16;
  store_conv(xout, b_dw, ax0, rbx, 0, colb);
  store_conv(xout, b_dw, ax1, rbx, 1, colb);
  store_conv(xout, b_dw, ax2, rbx, 2, colb);
  store_conv(xout, b_dw, ax3, rbx, 3, colb);
  store_conv(xout, b_dw, ay0, rby, 0, colb);
  store_conv(xout, b_dw, ay1, rby, 1, colb);
  store_conv(xout, b_dw, ay2, rby, 2, colb);
  store_conv(xout, b_dw, ay3, rby, 3, colb);
}

// ---------------- GroupNorm(1, C) per-batch sums ----------------
__global__ __launch_bounds__(256) void gn_stats_kernel(const float* __restrict__ x,
                                                       const int* __restrict__ bids,
                                                       float* __restrict__ stats) {
  __shared__ float s[6];
  int tid = threadIdx.x;
  if (tid < 6) s[tid] = 0.f;
  __syncthreads();
  int r = blockIdx.x * 256 + tid;
  if (r < NPTS) {
    const float4* xr = (const float4*)(x + (size_t)r * 64);
    float sm = 0.f, sq = 0.f;
#pragma unroll
    for (int i = 0; i < 16; ++i) {
      float4 v = xr[i];
      sm += v.x + v.y + v.z + v.w;
      sq += v.x * v.x + v.y * v.y + v.z * v.z + v.w * v.w;
    }
    int b = bids[r];
    atomicAdd(&s[b * 3 + 0], sm);
    atomicAdd(&s[b * 3 + 1], sq);
    atomicAdd(&s[b * 3 + 2], 64.f);
  }
  __syncthreads();
  if (tid < 6) atomicAdd(&stats[tid], s[tid]);
}

// ---------------- GN-normalize + pwconv1 + ReLU + GRN L2 accumulation ----------------
__global__ __launch_bounds__(64) void pw1_wmma(
    const float* __restrict__ x,
    const float* __restrict__ gng, const float* __restrict__ gnb,
    const uint4* __restrict__ wB1,
    const float* __restrict__ stats,
    const int* __restrict__ bids,
    unsigned short* __restrict__ h,               // [N, 256] bf16
    float* __restrict__ gx2)                      // [B, 256]
{
  const int tid  = threadIdx.x;
  const int lane = tid & 31;
  const int tile = blockIdx.x * PW_WAVES + (tid >> 5);
  const int half = lane >> 4;
  const int row  = tile * 16 + (lane & 15);
  const int b    = bids[tile * 16];

  float s0 = stats[b * 3 + 0], s1 = stats[b * 3 + 1], cn = stats[b * 3 + 2];
  float mean = s0 / cn;
  float rstd = rsqrtf(s1 / cn - mean * mean + 1e-6f);

  Frag a0, a1;
  const float* xr = x + (size_t)row * 64;
#pragma unroll
  for (int kc = 0; kc < 2; ++kc) {
    Frag& A = kc ? a1 : a0;
#pragma unroll
    for (int g = 0; g < 2; ++g) {
      int c = kc * 32 + half * 8 + g * 16;
      float4 xa = *(const float4*)(xr + c);
      float4 xb = *(const float4*)(xr + c + 4);
      float4 ga = *(const float4*)(gng + c);
      float4 gb = *(const float4*)(gng + c + 4);
      float4 ba = *(const float4*)(gnb + c);
      float4 bb = *(const float4*)(gnb + c + 4);
      A.u[g * 4 + 0] = pack2bf((xa.x - mean) * rstd * ga.x + ba.x, (xa.y - mean) * rstd * ga.y + ba.y);
      A.u[g * 4 + 1] = pack2bf((xa.z - mean) * rstd * ga.z + ba.z, (xa.w - mean) * rstd * ga.w + ba.w);
      A.u[g * 4 + 2] = pack2bf((xb.x - mean) * rstd * gb.x + bb.x, (xb.y - mean) * rstd * gb.y + bb.y);
      A.u[g * 4 + 3] = pack2bf((xb.z - mean) * rstd * gb.z + bb.z, (xb.w - mean) * rstd * gb.w + bb.w);
    }
  }

  v8f vzero = {};
  v8f acc[16];
#pragma unroll
  for (int nt = 0; nt < 16; ++nt) acc[nt] = vzero;

#pragma unroll
  for (int nt = 0; nt < 16; ++nt) {
    Frag bfa, bfb;                                 // distinct regs -> loads clause up
    const uint4* p0 = wB1 + (nt * 2 + 0) * 64 + lane * 2;
    const uint4* p1 = wB1 + (nt * 2 + 1) * 64 + lane * 2;
    bfa.q[0] = p0[0]; bfa.q[1] = p0[1];
    bfb.q[0] = p1[0]; bfb.q[1] = p1[1];
    WMMA_BF16(a0, bfa, acc[nt]);
    WMMA_BF16(a1, bfb, acc[nt]);
  }

  const int colb = lane & 15;
  const int rb = tile * 16 + half * 8;
#pragma unroll
  for (int nt = 0; nt < 16; ++nt) {
    int col = nt * 16 + colb;
    float part = 0.f;
#pragma unroll
    for (int j = 0; j < 8; ++j) {
      float v = fmaxf(acc[nt][j], 0.f);
      h[(size_t)(rb + j) * 256 + col] = f2bf(v);
      part += v * v;
    }
    part += __shfl_xor(part, 16, 32);
    if (half == 0) atomicAdd(&gx2[b * 256 + col], part);
  }
}

// ---------------- GRN: scale[b][c] = grn_gamma[c] * nx + 1 ----------------
__global__ void grn_scale_kernel(const float* __restrict__ gx2, const float* __restrict__ grng,
                                 float* __restrict__ scale) {
  __shared__ float sum[2];
  int t = threadIdx.x;           // 512 = B * 256
  int b = t >> 8, c = t & 255;
  if (t < 2) sum[t] = 0.f;
  __syncthreads();
  float gx = sqrtf(gx2[t]);
  atomicAdd(&sum[b], gx);
  __syncthreads();
  float nx = gx / (sum[b] * (1.f / 256.f) + 1e-6f);
  scale[t] = grng[c] * nx + 1.f;
}

// ---------------- GRN apply + pwconv2 + residual ----------------
__device__ __forceinline__ void store_res(float* __restrict__ out, const float* __restrict__ feats,
                                          v8f a, int rb, int nt, int colb) {
#pragma unroll
  for (int j = 0; j < 8; ++j) {
    size_t o = (size_t)(rb + j) * 64 + nt * 16 + colb;
    out[o] = feats[o] + a[j];
  }
}

__global__ __launch_bounds__(64) void pw2_wmma(
    const unsigned short* __restrict__ h,
    const float* __restrict__ scale,
    const float* __restrict__ grnb,
    const uint4* __restrict__ wB2,
    const float* __restrict__ feats,
    const int* __restrict__ bids,
    float* __restrict__ out)
{
  const int tid  = threadIdx.x;
  const int lane = tid & 31;
  const int tile = blockIdx.x * PW_WAVES + (tid >> 5);
  const int half = lane >> 4;
  const int row  = tile * 16 + (lane & 15);
  const int b    = bids[tile * 16];
  const unsigned int* hrow = (const unsigned int*)(h + (size_t)row * 256);
  const float* sc = scale + b * 256;

  v8f acc0 = {}, acc1 = {}, acc2 = {}, acc3 = {};

#pragma unroll
  for (int kc = 0; kc < 8; ++kc) {
    Frag a;
#pragma unroll
    for (int g = 0; g < 2; ++g) {
      int c = kc * 32 + half * 8 + g * 16;
      uint4  hv = *(const uint4*)(hrow + (c >> 1));
      float4 sa = *(const float4*)(sc + c);
      float4 sb = *(const float4*)(sc + c + 4);
      float4 ta = *(const float4*)(grnb + c);
      float4 tb = *(const float4*)(grnb + c + 4);
      a.u[g * 4 + 0] = pack2bf(bflo(hv.x) * sa.x + ta.x, bfhi(hv.x) * sa.y + ta.y);
      a.u[g * 4 + 1] = pack2bf(bflo(hv.y) * sa.z + ta.z, bfhi(hv.y) * sa.w + ta.w);
      a.u[g * 4 + 2] = pack2bf(bflo(hv.z) * sb.x + tb.x, bfhi(hv.z) * sb.y + tb.y);
      a.u[g * 4 + 3] = pack2bf(bflo(hv.w) * sb.z + tb.z, bfhi(hv.w) * sb.w + tb.w);
    }
    Frag f0, f1, f2, f3;                           // distinct regs -> loads clause up
    const uint4* p0 = wB2 + (0 * 8 + kc) * 64 + lane * 2;
    const uint4* p1 = wB2 + (1 * 8 + kc) * 64 + lane * 2;
    const uint4* p2 = wB2 + (2 * 8 + kc) * 64 + lane * 2;
    const uint4* p3 = wB2 + (3 * 8 + kc) * 64 + lane * 2;
    f0.q[0] = p0[0]; f0.q[1] = p0[1];
    f1.q[0] = p1[0]; f1.q[1] = p1[1];
    f2.q[0] = p2[0]; f2.q[1] = p2[1];
    f3.q[0] = p3[0]; f3.q[1] = p3[1];
    WMMA_BF16(a, f0, acc0);
    WMMA_BF16(a, f1, acc1);
    WMMA_BF16(a, f2, acc2);
    WMMA_BF16(a, f3, acc3);
  }

  const int colb = lane & 15;
  const int rb = tile * 16 + half * 8;
  store_res(out, feats, acc0, rb, 0, colb);
  store_res(out, feats, acc1, rb, 1, colb);
  store_res(out, feats, acc2, rb, 2, colb);
  store_res(out, feats, acc3, rb, 3, colb);
}

// ---------------- host launch ----------------
extern "C" void kernel_launch(void* const* d_in, const int* in_sizes, int n_in,
                              void* d_out, int out_size, void* d_ws, size_t ws_size,
                              hipStream_t stream) {
  const float* feats  = (const float*)d_in[0];
  const float* w_dw   = (const float*)d_in[1];
  const float* b_dw   = (const float*)d_in[2];
  const float* gng    = (const float*)d_in[3];
  const float* gnb    = (const float*)d_in[4];
  const float* w_pw1  = (const float*)d_in[5];
  const float* grng   = (const float*)d_in[6];
  const float* grnb   = (const float*)d_in[7];
  const float* w_pw2  = (const float*)d_in[8];
  const int*   nbr    = (const int*)d_in[9];
  const int*   bids   = (const int*)d_in[10];
  float* out = (float*)d_out;

  char* ws = (char*)d_ws;
  unsigned short* featsbf = (unsigned short*)(ws + OFF_FEATSBF);
  unsigned int*   wBdw    = (unsigned int*)(ws + OFF_WBDW);
  unsigned int*   wB1     = (unsigned int*)(ws + OFF_WB1);
  unsigned int*   wB2     = (unsigned int*)(ws + OFF_WB2);
  float*          xbuf    = (float*)(ws + OFF_X);
  unsigned short* hbuf    = (unsigned short*)(ws + OFF_H);
  float*          stats   = (float*)(ws + OFF_STATS);
  float*          gx2     = (float*)(ws + OFF_GX2);
  float*          scl     = (float*)(ws + OFF_SCALE);
  unsigned int*   rowN    = (unsigned int*)featsbf + (size_t)NPTS * 32; // zero sentinel row

  init_kernel<<<1, 1024, 0, stream>>>(stats, gx2, rowN);

  const int feat_dw = NPTS * 32;                                  // 3.2M packed dwords
  cvt_feats<<<(feat_dw + 255) / 256, 256, 0, stream>>>(feats, (unsigned int*)featsbf, feat_dw);

  const int dw_total  = KKOFF * 2048;                             // 702,464 dwords
  pack_w<<<(dw_total + 255) / 256, 256, 0, stream>>>(w_dw, wBdw, 64, 64, dw_total);
  pack_w<<<32, 256, 0, stream>>>(w_pw1, wB1, 64, 256, 8192);
  pack_w<<<32, 256, 0, stream>>>(w_pw2, wB2, 256, 64, 8192);

  conv_dw_wmma<<<CONV_BLOCKS, 512, 0, stream>>>(featsbf, (const uint4*)wBdw, nbr, b_dw, xbuf);

  gn_stats_kernel<<<(NPTS + 255) / 256, 256, 0, stream>>>(xbuf, bids, stats);

  pw1_wmma<<<PW_BLOCKS, 64, 0, stream>>>(xbuf, gng, gnb, (const uint4*)wB1, stats, bids, hbuf, gx2);

  grn_scale_kernel<<<1, 512, 0, stream>>>(gx2, grng, scl);

  pw2_wmma<<<PW_BLOCKS, 64, 0, stream>>>(hbuf, scl, grnb, (const uint4*)wB2, feats, bids, out);
}
